// APPNP_15625091022891
// MI455X (gfx1250) — matrix operand order repaired and checked by reading, not compiled, not run on hardware
//
#include <hip/hip_runtime.h>
#include <hip/hip_bf16.h>

// APPNP on MI455X (gfx1250).
// Dominant op: 10x dense PPR steps  x_cl = 0.9*(adj @ x_cl) + 0.1*h0,
// adj = 16384^2. Memory-bound on adj -> pre-convert adj to bf16 once
// (10.7 GB -> ~6.9 GB total HBM traffic) and run the GEMM with
// v_wmma_f32_16x16x32_bf16 (wave32). A tiles move HBM->LDS with
// global_load_async_to_lds_b128, double-buffered so tile t+1 copies
// while tile t is in the WMMAs (ASYNCcnt pipelining).

typedef __attribute__((ext_vector_type(16))) __bf16          v16bf;
typedef __attribute__((ext_vector_type(8)))  float           v8f;
typedef __attribute__((ext_vector_type(8)))  unsigned short  ushort8;
typedef __attribute__((ext_vector_type(4)))  int             v4i;

typedef __attribute__((address_space(1))) v4i gv4i;   // global int4
typedef __attribute__((address_space(3))) v4i lv4i;   // LDS int4

static constexpr int N_NODES = 16384;
static constexpr int NFEAT   = 512;
static constexpr int NHID    = 64;
static constexpr int NCLASS  = 40;
static constexpr int KSTEPS  = 10;

#define M_TILE 128
#define K_TILE 32
#define KPAD   (K_TILE + 8)

#if defined(__HIP_DEVICE_COMPILE__) && \
    __has_builtin(__builtin_amdgcn_global_load_async_to_lds_b128) && \
    __has_builtin(__builtin_amdgcn_s_wait_asynccnt)
#define USE_ASYNC_LDS 1
#else
#define USE_ASYNC_LDS 0
#endif

__device__ __forceinline__ __bf16 tobf(float f) { return (__bf16)f; }  // v_cvt_pk_bf16_f32

union AFrag { v16bf v; __bf16 s[16]; };
union BFrag { v16bf v; ushort8 h[2]; };

// ---------------------------------------------------------------------------
// Double-buffered GEMM, A already bf16 (row-major), B fp32 [K x 64].
// MODE 0: relu epilogue; MODE 1: 0.9*acc + 0.1*h0 epilogue.
// ---------------------------------------------------------------------------
template <int MODE>
__global__ __launch_bounds__(256)
void gemm_n64_bf16A(const __bf16* __restrict__ A, const float* __restrict__ B,
                    const float* __restrict__ h0, float* __restrict__ out, int K) {
  __shared__ __bf16 sA [2][M_TILE][KPAD];   // 2 x 10 KB
  __shared__ __bf16 sBT[2][64]    [KPAD];   // 2 x 5 KB, transposed B

  const int tid  = threadIdx.x;
  const int wave = tid >> 5;
  const int lane = tid & 31;
  const int m0   = blockIdx.x * M_TILE;

  v8f acc0 = {}, acc1 = {}, acc2 = {}, acc3 = {};

  // A cooperative-load coords: 128 rows x 32 bf16, 32 B per thread
  const int rowA_ld = tid >> 1;
  const int colA_ld = (tid & 1) * 16;
  // B cooperative-load coords: 2 consecutive K-rows x 4 cols per thread
  const int kpair = (tid >> 4) * 2;   // 0,2,..,30
  const int cbase = (tid & 15) * 4;   // 0,4,..,60

  // fragment coords (this code's contiguous k order: element s of lane half h is k = 16h + s)
  const int fragRow = wave * 16 + (lane & 15);
  const int klo = (lane < 16) ? 0 : 16;  // A: lanes 0-15 -> K 0-15
  const int kB  = (lane < 16) ? 0 : 16;  // B: lanes 0-15 -> K 0-15
  const int n   = lane & 15;

  const int nTiles = K / K_TILE;

  // ---- prologue: stage tile 0 into buffer 0 ----
  {
    const __bf16* ap = A + (size_t)(m0 + rowA_ld) * K + colA_ld;
#if USE_ASYNC_LDS
    __builtin_amdgcn_global_load_async_to_lds_b128(
        (gv4i*)ap,       (lv4i*)&sA[0][rowA_ld][colA_ld],     0, 0);
    __builtin_amdgcn_global_load_async_to_lds_b128(
        (gv4i*)(ap + 8), (lv4i*)&sA[0][rowA_ld][colA_ld + 8], 0, 0);
#else
    *(ushort8*)&sA[0][rowA_ld][colA_ld]     = *(const ushort8*)ap;
    *(ushort8*)&sA[0][rowA_ld][colA_ld + 8] = *(const ushort8*)(ap + 8);
#endif
    const float* bp = B + (size_t)kpair * 64 + cbase;
    float4 bl = *(const float4*)bp;
    float4 bh = *(const float4*)(bp + 64);
    const float* blp = (const float*)&bl;
    const float* bhp = (const float*)&bh;
#pragma unroll
    for (int i = 0; i < 4; ++i) {
      union { __bf16 b[2]; unsigned int u; } p;
      p.b[0] = tobf(blp[i]); p.b[1] = tobf(bhp[i]);
      *(unsigned int*)&sBT[0][cbase + i][kpair] = p.u;
    }
#if USE_ASYNC_LDS
    __builtin_amdgcn_s_wait_asynccnt(0);
#endif
    __syncthreads();
  }

  for (int t = 0; t < nTiles; ++t) {
    const int cur = t & 1;
    const int nxt = cur ^ 1;
    const bool has_next = (t + 1) < nTiles;
    const int knext = (t + 1) * K_TILE;

    float4 bl = {}, bh = {};
#if !USE_ASYNC_LDS
    ushort8 av0 = {}, av1 = {};
#endif
    if (has_next) {
      // issue tile t+1: A async HBM->LDS, B global->regs (consumed post-WMMA)
      const __bf16* ap = A + (size_t)(m0 + rowA_ld) * K + (knext + colA_ld);
#if USE_ASYNC_LDS
      __builtin_amdgcn_global_load_async_to_lds_b128(
          (gv4i*)ap,       (lv4i*)&sA[nxt][rowA_ld][colA_ld],     0, 0);
      __builtin_amdgcn_global_load_async_to_lds_b128(
          (gv4i*)(ap + 8), (lv4i*)&sA[nxt][rowA_ld][colA_ld + 8], 0, 0);
#else
      av0 = *(const ushort8*)ap;
      av1 = *(const ushort8*)(ap + 8);
#endif
      __builtin_prefetch(ap + K_TILE, 0, 1);   // tile t+2 -> L2
      const float* bp = B + (size_t)(knext + kpair) * 64 + cbase;
      bl = *(const float4*)bp;
      bh = *(const float4*)(bp + 64);
    }

    // ---- compute tile t: aligned 16B LDS frag loads + 4 WMMAs ----
    BFrag af;
    af.h[0] = *(const ushort8*)&sA[cur][fragRow][klo];
    af.h[1] = *(const ushort8*)&sA[cur][fragRow][klo + 8];

    BFrag bf0, bf1, bf2, bf3;
    bf0.h[0] = *(const ushort8*)&sBT[cur][n +  0][kB];
    bf0.h[1] = *(const ushort8*)&sBT[cur][n +  0][kB + 8];
    bf1.h[0] = *(const ushort8*)&sBT[cur][n + 16][kB];
    bf1.h[1] = *(const ushort8*)&sBT[cur][n + 16][kB + 8];
    bf2.h[0] = *(const ushort8*)&sBT[cur][n + 32][kB];
    bf2.h[1] = *(const ushort8*)&sBT[cur][n + 32][kB + 8];
    bf3.h[0] = *(const ushort8*)&sBT[cur][n + 48][kB];
    bf3.h[1] = *(const ushort8*)&sBT[cur][n + 48][kB + 8];

    acc0 = __builtin_amdgcn_wmma_f32_16x16x32_bf16(false, af.v, false, bf0.v,
                                                   (short)0, acc0, false, false);
    acc1 = __builtin_amdgcn_wmma_f32_16x16x32_bf16(false, af.v, false, bf1.v,
                                                   (short)0, acc1, false, false);
    acc2 = __builtin_amdgcn_wmma_f32_16x16x32_bf16(false, af.v, false, bf2.v,
                                                   (short)0, acc2, false, false);
    acc3 = __builtin_amdgcn_wmma_f32_16x16x32_bf16(false, af.v, false, bf3.v,
                                                   (short)0, acc3, false, false);

    if (has_next) {
      // stage B(t+1): packed (k,k+1) pairs -> single b32 DS stores
      const float* blp = (const float*)&bl;
      const float* bhp = (const float*)&bh;
#pragma unroll
      for (int i = 0; i < 4; ++i) {
        union { __bf16 b[2]; unsigned int u; } p;
        p.b[0] = tobf(blp[i]); p.b[1] = tobf(bhp[i]);
        *(unsigned int*)&sBT[nxt][cbase + i][kpair] = p.u;
      }
#if !USE_ASYNC_LDS
      *(ushort8*)&sA[nxt][rowA_ld][colA_ld]     = av0;
      *(ushort8*)&sA[nxt][rowA_ld][colA_ld + 8] = av1;
#endif
    }
#if USE_ASYNC_LDS
    __builtin_amdgcn_s_wait_asynccnt(0);   // A(t+1) landed in LDS
#endif
    __syncthreads();
  }

  // ---- epilogue: C/D layout (lane = N + 16*(M>=8), vgpr j = M%8) ----
  const int outRow0 = m0 + wave * 16 + ((lane >> 4) << 3);
#pragma unroll
  for (int j = 0; j < 8; ++j) {
    const int r = outRow0 + j;
    float* orow = out + (size_t)r * 64;
    float v0 = acc0[j], v1 = acc1[j], v2 = acc2[j], v3 = acc3[j];
    if (MODE == 0) {
      orow[n +  0] = fmaxf(v0, 0.0f);
      orow[n + 16] = fmaxf(v1, 0.0f);
      orow[n + 32] = fmaxf(v2, 0.0f);
      orow[n + 48] = fmaxf(v3, 0.0f);
    } else {
      const float* hrow = h0 + (size_t)r * 64;
      orow[n +  0] = 0.9f * v0 + 0.1f * hrow[n +  0];
      orow[n + 16] = 0.9f * v1 + 0.1f * hrow[n + 16];
      orow[n + 32] = 0.9f * v2 + 0.1f * hrow[n + 32];
      orow[n + 48] = 0.9f * v3 + 0.1f * hrow[n + 48];
    }
  }
}

// ---------------------------------------------------------------------------
// Fallback GEMM: A fp32 (used only if workspace can't hold bf16 adj).
// ---------------------------------------------------------------------------
template <int MODE>
__global__ __launch_bounds__(256)
void gemm_n64_f32A(const float* __restrict__ A, const float* __restrict__ B,
                   const float* __restrict__ h0, float* __restrict__ out, int K) {
  __shared__ float  sA [M_TILE][K_TILE + 4];
  __shared__ __bf16 sBT[64]    [KPAD];

  const int tid  = threadIdx.x;
  const int wave = tid >> 5;
  const int lane = tid & 31;
  const int m0   = blockIdx.x * M_TILE;

  v8f acc0 = {}, acc1 = {}, acc2 = {}, acc3 = {};

  const int rowA_ld = tid >> 1;
  const int colA_ld = (tid & 1) * 16;
  const int kpair = (tid >> 4) * 2;
  const int cbase = (tid & 15) * 4;

  const int fragRow = wave * 16 + (lane & 15);
  const int klo = (lane < 16) ? 0 : 16;
  const int kB  = (lane < 16) ? 0 : 16;
  const int n   = lane & 15;

  for (int k0 = 0; k0 < K; k0 += K_TILE) {
    const float* ap = A + (size_t)(m0 + rowA_ld) * K + (k0 + colA_ld);
    float4 a0 = *(const float4*)(ap + 0);
    float4 a1 = *(const float4*)(ap + 4);
    float4 a2 = *(const float4*)(ap + 8);
    float4 a3 = *(const float4*)(ap + 12);
    if (k0 + K_TILE < K) __builtin_prefetch(ap + K_TILE, 0, 1);
    *(float4*)&sA[rowA_ld][colA_ld + 0]  = a0;
    *(float4*)&sA[rowA_ld][colA_ld + 4]  = a1;
    *(float4*)&sA[rowA_ld][colA_ld + 8]  = a2;
    *(float4*)&sA[rowA_ld][colA_ld + 12] = a3;

    const float* bp = B + (size_t)(k0 + kpair) * 64 + cbase;
    float4 bl = *(const float4*)bp;
    float4 bh = *(const float4*)(bp + 64);
    const float* blp = (const float*)&bl;
    const float* bhp = (const float*)&bh;
#pragma unroll
    for (int i = 0; i < 4; ++i) {
      union { __bf16 b[2]; unsigned int u; } p;
      p.b[0] = tobf(blp[i]); p.b[1] = tobf(bhp[i]);
      *(unsigned int*)&sBT[cbase + i][kpair] = p.u;
    }
    __syncthreads();

    AFrag af;
    const float* arow = &sA[fragRow][0];
#pragma unroll
    for (int i = 0; i < 8; ++i) af.s[i]     = tobf(arow[klo + i]);
#pragma unroll
    for (int i = 0; i < 8; ++i) af.s[8 + i] = tobf(arow[klo + 8 + i]);

    BFrag bf0, bf1, bf2, bf3;
    bf0.h[0] = *(const ushort8*)&sBT[n +  0][kB];
    bf0.h[1] = *(const ushort8*)&sBT[n +  0][kB + 8];
    bf1.h[0] = *(const ushort8*)&sBT[n + 16][kB];
    bf1.h[1] = *(const ushort8*)&sBT[n + 16][kB + 8];
    bf2.h[0] = *(const ushort8*)&sBT[n + 32][kB];
    bf2.h[1] = *(const ushort8*)&sBT[n + 32][kB + 8];
    bf3.h[0] = *(const ushort8*)&sBT[n + 48][kB];
    bf3.h[1] = *(const ushort8*)&sBT[n + 48][kB + 8];

    acc0 = __builtin_amdgcn_wmma_f32_16x16x32_bf16(false, af.v, false, bf0.v,
                                                   (short)0, acc0, false, false);
    acc1 = __builtin_amdgcn_wmma_f32_16x16x32_bf16(false, af.v, false, bf1.v,
                                                   (short)0, acc1, false, false);
    acc2 = __builtin_amdgcn_wmma_f32_16x16x32_bf16(false, af.v, false, bf2.v,
                                                   (short)0, acc2, false, false);
    acc3 = __builtin_amdgcn_wmma_f32_16x16x32_bf16(false, af.v, false, bf3.v,
                                                   (short)0, acc3, false, false);
    __syncthreads();
  }

  const int outRow0 = m0 + wave * 16 + ((lane >> 4) << 3);
#pragma unroll
  for (int j = 0; j < 8; ++j) {
    const int r = outRow0 + j;
    float* orow = out + (size_t)r * 64;
    float v0 = acc0[j], v1 = acc1[j], v2 = acc2[j], v3 = acc3[j];
    if (MODE == 0) {
      orow[n +  0] = fmaxf(v0, 0.0f);
      orow[n + 16] = fmaxf(v1, 0.0f);
      orow[n + 32] = fmaxf(v2, 0.0f);
      orow[n + 48] = fmaxf(v3, 0.0f);
    } else {
      const float* hrow = h0 + (size_t)r * 64;
      orow[n +  0] = 0.9f * v0 + 0.1f * hrow[n +  0];
      orow[n + 16] = 0.9f * v1 + 0.1f * hrow[n + 16];
      orow[n + 32] = 0.9f * v2 + 0.1f * hrow[n + 32];
      orow[n + 48] = 0.9f * v3 + 0.1f * hrow[n + 48];
    }
  }
}

// fp32 -> bf16 bulk convert (n multiple of 8)
__global__ __launch_bounds__(256)
void f32_to_bf16(const float* __restrict__ src, __bf16* __restrict__ dst,
                 long long n) {
  long long i      = ((long long)blockIdx.x * 256 + threadIdx.x) * 8;
  long long stride = (long long)gridDim.x * 256 * 8;
  for (; i < n; i += stride) {
    float4 f0 = *(const float4*)(src + i);
    float4 f1 = *(const float4*)(src + i + 4);
    union { __bf16 b[8]; ushort8 u; } o;
    o.b[0] = tobf(f0.x); o.b[1] = tobf(f0.y);
    o.b[2] = tobf(f0.z); o.b[3] = tobf(f0.w);
    o.b[4] = tobf(f1.x); o.b[5] = tobf(f1.y);
    o.b[6] = tobf(f1.z); o.b[7] = tobf(f1.w);
    *(ushort8*)(dst + i) = o.u;
  }
}

// out[r, :] = log_softmax(x_cl[r, :] @ W2)
__global__ __launch_bounds__(256)
void classify_logsoftmax(const float* __restrict__ xcl,
                         const float* __restrict__ W2,
                         float* __restrict__ out) {
  __shared__ float sW[NHID * NCLASS];
  for (int i = threadIdx.x; i < NHID * NCLASS; i += 256) sW[i] = W2[i];
  __syncthreads();

  const int r = blockIdx.x * 256 + threadIdx.x;
  const float* xr = xcl + (size_t)r * NHID;

  float h[NCLASS];
#pragma unroll
  for (int c = 0; c < NCLASS; ++c) h[c] = 0.0f;
  for (int k = 0; k < NHID; ++k) {
    const float xv = xr[k];
    const float* wrow = &sW[k * NCLASS];
#pragma unroll
    for (int c = 0; c < NCLASS; ++c) h[c] = fmaf(xv, wrow[c], h[c]);
  }
  float m = h[0];
#pragma unroll
  for (int c = 1; c < NCLASS; ++c) m = fmaxf(m, h[c]);
  float s = 0.0f;
#pragma unroll
  for (int c = 0; c < NCLASS; ++c) s += __expf(h[c] - m);
  const float ls = __logf(s);
  float* orow = out + (size_t)r * NCLASS;
#pragma unroll
  for (int c = 0; c < NCLASS; ++c) orow[c] = h[c] - m - ls;
}

extern "C" void kernel_launch(void* const* d_in, const int* in_sizes, int n_in,
                              void* d_out, int out_size, void* d_ws, size_t ws_size,
                              hipStream_t stream) {
  const float* x   = (const float*)d_in[0];   // [16384, 512]
  const float* adj = (const float*)d_in[1];   // [16384, 16384]
  const float* W1  = (const float*)d_in[2];   // [512, 64]
  const float* W2  = (const float*)d_in[3];   // [64, 40]
  float* out = (float*)d_out;                 // log_softmax ++ x_cl

  float* xcl_out = out + (size_t)N_NODES * NCLASS;

  const size_t adj_elems = (size_t)N_NODES * N_NODES;
  const size_t x_elems   = (size_t)N_NODES * NFEAT;
  const size_t hid_elems = (size_t)N_NODES * NHID;
  const size_t need_bf16 = adj_elems * 2 + x_elems * 2 + 3 * hid_elems * 4;

  dim3 block(256);
  dim3 grid(N_NODES / M_TILE);
  dim3 cls_grid(N_NODES / 256);

  if (ws_size >= need_bf16) {
    // --- bf16-staged path: adj converted once, read 10x at half the bytes ---
    __bf16* adjbf = (__bf16*)d_ws;
    __bf16* xbf   = adjbf + adj_elems;
    float*  h0    = (float*)(xbf + x_elems);
    float*  ping  = h0 + hid_elems;
    float*  pong  = ping + hid_elems;

    f32_to_bf16<<<dim3(4096), block, 0, stream>>>(adj, adjbf, (long long)adj_elems);
    f32_to_bf16<<<dim3(512),  block, 0, stream>>>(x,   xbf,   (long long)x_elems);

    gemm_n64_bf16A<0><<<grid, block, 0, stream>>>(xbf, W1, nullptr, h0, NFEAT);

    const float* cur = h0;
    for (int it = 0; it < KSTEPS; ++it) {
      float* dst = (it == KSTEPS - 1) ? xcl_out : ((it & 1) ? pong : ping);
      gemm_n64_bf16A<1><<<grid, block, 0, stream>>>(adjbf, cur, h0, dst, N_NODES);
      cur = dst;
    }
    classify_logsoftmax<<<cls_grid, block, 0, stream>>>(cur, W2, out);
  } else {
    // --- fallback: stream fp32 adj, convert in-kernel ---
    float* h0   = (float*)d_ws;
    float* ping = h0 + hid_elems;
    float* pong = ping + hid_elems;

    gemm_n64_f32A<0><<<grid, block, 0, stream>>>(x, W1, nullptr, h0, NFEAT);

    const float* cur = h0;
    for (int it = 0; it < KSTEPS; ++it) {
      float* dst = (it == KSTEPS - 1) ? xcl_out : ((it & 1) ? pong : ping);
      gemm_n64_f32A<1><<<grid, block, 0, stream>>>(adj, cur, h0, dst, N_NODES);
      cur = dst;
    }
    classify_logsoftmax<<<cls_grid, block, 0, stream>>>(cur, W2, out);
  }
}
